// CrossAttention_86114094285429
// MI455X (gfx1250) — compile-verified, matrix operand order checked by reading, not run on hardware
//
#include <hip/hip_runtime.h>
#include <hip/hip_bf16.h>

typedef __attribute__((ext_vector_type(16))) _Float16 v16h;
typedef __attribute__((ext_vector_type(8)))  _Float16 v8h;
typedef __attribute__((ext_vector_type(8)))  float    v8f;

#define N_NODES 2048
#define DHEAD   256
#define PDIM    2048   // H * D
#define SCALE_F 0.0625f  // 256^-0.5

// Load a 16-bit WMMA fragment (A of 16x32, or B of 32x16 for A@B^T) for one lane.
// Caller passes p = &mat[(rowBase + (lane&15)) * stride + colBase + (lane>>4)*8 + k].
// Per ISA layout: elements 0..7 hold K = koff+0..7, elements 8..15 hold K = koff+16..23.
__device__ __forceinline__ v16h load_frag(const _Float16* __restrict__ p) {
    v8h lo = *(const v8h*)(p);
    v8h hi = *(const v8h*)(p + 16);
    v16h f;
#pragma unroll
    for (int i = 0; i < 8; ++i) { f[i] = lo[i]; f[8 + i] = hi[i]; }
    return f;
}

// ---------------- Stage 0a: convert embeddings f32 -> f16 ----------------
__global__ void cvt_emb_kernel(const float* __restrict__ e1, const float* __restrict__ e2,
                               _Float16* __restrict__ o1, _Float16* __restrict__ o2) {
    int idx = blockIdx.x * blockDim.x + threadIdx.x;   // 0 .. 2048*256-1
    const float* src = blockIdx.z ? e2 : e1;
    _Float16*    dst = blockIdx.z ? o2 : o1;
    dst[idx] = (_Float16)src[idx];
}

// ---------------- Stage 0b: transpose + convert W [256,2048] -> Wt [2048,256] f16 ----
__global__ void transpose_w_kernel(const float* __restrict__ Wq, const float* __restrict__ Wk,
                                   _Float16* __restrict__ Wqt, _Float16* __restrict__ Wkt) {
    __shared__ float tile[32][33];
    const float* W  = blockIdx.z ? Wk  : Wq;
    _Float16*    Wt = blockIdx.z ? Wkt : Wqt;
    int j0 = blockIdx.x * 32;   // column of W (0..2047)
    int k0 = blockIdx.y * 32;   // row of W    (0..255)
    int tx = threadIdx.x, ty = threadIdx.y;
    tile[ty][tx] = W[(size_t)(k0 + ty) * PDIM + j0 + tx];
    __syncthreads();
    Wt[(size_t)(j0 + ty) * DHEAD + k0 + tx] = (_Float16)tile[tx][ty];
}

// ---------------- Stage 1: projection GEMM  P = emb_f16 @ Wt^T + b  (f16 out) -------
// Block tile 128x128, 8 waves (2 in M x 4 in N), wave tile 64x32 = 4x2 WMMA frags.
__global__ void proj_kernel(const _Float16* __restrict__ e1, const _Float16* __restrict__ e2,
                            const _Float16* __restrict__ Wqt, const _Float16* __restrict__ Wkt,
                            const float* __restrict__ bq, const float* __restrict__ bk,
                            _Float16* __restrict__ Q1, _Float16* __restrict__ Q2,
                            _Float16* __restrict__ K1, _Float16* __restrict__ K2) {
    int z = blockIdx.z;                       // 0:Q1 1:Q2 2:K1 3:K2
    const _Float16* A = (z & 1) ? e2  : e1;
    const _Float16* B = (z < 2) ? Wqt : Wkt;
    const float* bias = (z < 2) ? bq  : bk;
    _Float16* P = (z == 0) ? Q1 : (z == 1) ? Q2 : (z == 2) ? K1 : K2;

    int lane  = threadIdx.x & 31;
    int w     = threadIdx.x >> 5;
    int waveM = w & 1, waveN = w >> 1;
    int rowBase = blockIdx.y * 128 + waveM * 64;
    int colBase = blockIdx.x * 128 + waveN * 32;
    int lrow = lane & 15;
    int koff = (lane >> 4) * 8;

    const _Float16* aP = A + (size_t)(rowBase + lrow) * DHEAD + koff;
    const _Float16* bP = B + (size_t)(colBase + lrow) * DHEAD + koff;

    v8f acc[4][2] = {};
#pragma unroll
    for (int k = 0; k < DHEAD; k += 32) {
        v16h af[4], bf[2];
#pragma unroll
        for (int mi = 0; mi < 4; ++mi) af[mi] = load_frag(aP + (size_t)mi * 16 * DHEAD + k);
#pragma unroll
        for (int ni = 0; ni < 2; ++ni) bf[ni] = load_frag(bP + (size_t)ni * 16 * DHEAD + k);
#pragma unroll
        for (int mi = 0; mi < 4; ++mi)
#pragma unroll
            for (int ni = 0; ni < 2; ++ni)
                acc[mi][ni] = __builtin_amdgcn_wmma_f32_16x16x32_f16(
                    false, af[mi], false, bf[ni], (short)0, acc[mi][ni], false, false);
    }

    int j0 = colBase + (lane & 15);
    int rowHalf = (lane >> 4) * 8;
#pragma unroll
    for (int ni = 0; ni < 2; ++ni) {
        float bj = bias[j0 + 16 * ni];
#pragma unroll
        for (int mi = 0; mi < 4; ++mi)
#pragma unroll
            for (int r = 0; r < 8; ++r) {
                int i = rowBase + mi * 16 + rowHalf + r;
                P[(size_t)i * PDIM + j0 + 16 * ni] = (_Float16)(acc[mi][ni][r] + bj);
            }
    }
}

// ---------------- Stage 2: masked attention scores ----------------------------------
// g in [0,16): g<8  -> a_1 head g : A=Q1, B=K2, mask t1[i]==t2[j]
//              g>=8 -> a_2 head g-8: A=K1, B=Q2, mask t2[i]==t1[j]
__global__ void attn_kernel(const _Float16* __restrict__ Q1, const _Float16* __restrict__ Q2,
                            const _Float16* __restrict__ K1, const _Float16* __restrict__ K2,
                            const int* __restrict__ t1, const int* __restrict__ t2,
                            float* __restrict__ out) {
    int g = blockIdx.z;
    int h = g & 7;
    bool first = (g < 8);
    const _Float16* A = first ? Q1 : K1;
    const _Float16* B = first ? K2 : Q2;
    const int* tr = first ? t1 : t2;
    const int* tc = first ? t2 : t1;
    int hoff = h * DHEAD;

    int lane  = threadIdx.x & 31;
    int w     = threadIdx.x >> 5;
    int waveM = w & 1, waveN = w >> 1;
    int rowBase = blockIdx.y * 128 + waveM * 64;
    int colBase = blockIdx.x * 128 + waveN * 32;
    int lrow = lane & 15;
    int koff = (lane >> 4) * 8;

    const _Float16* aP = A + (size_t)(rowBase + lrow) * PDIM + hoff + koff;
    const _Float16* bP = B + (size_t)(colBase + lrow) * PDIM + hoff + koff;

    v8f acc[4][2] = {};
#pragma unroll
    for (int k = 0; k < DHEAD; k += 32) {
        v16h af[4], bf[2];
#pragma unroll
        for (int mi = 0; mi < 4; ++mi) af[mi] = load_frag(aP + (size_t)mi * 16 * PDIM + k);
#pragma unroll
        for (int ni = 0; ni < 2; ++ni) bf[ni] = load_frag(bP + (size_t)ni * 16 * PDIM + k);
#pragma unroll
        for (int mi = 0; mi < 4; ++mi)
#pragma unroll
            for (int ni = 0; ni < 2; ++ni)
                acc[mi][ni] = __builtin_amdgcn_wmma_f32_16x16x32_f16(
                    false, af[mi], false, bf[ni], (short)0, acc[mi][ni], false, false);
    }

    float* og = out + (size_t)g * N_NODES * N_NODES;
    int j0 = colBase + (lane & 15);
    int rowHalf = (lane >> 4) * 8;
    int tj0 = tc[j0];
    int tj1 = tc[j0 + 16];
#pragma unroll
    for (int mi = 0; mi < 4; ++mi)
#pragma unroll
        for (int r = 0; r < 8; ++r) {
            int i  = rowBase + mi * 16 + rowHalf + r;
            int ti = tr[i];
            float v0 = acc[mi][0][r] * SCALE_F;
            float v1 = acc[mi][1][r] * SCALE_F;
            og[(size_t)i * N_NODES + j0]      = (ti == tj0) ? v0 : 0.0f;
            og[(size_t)i * N_NODES + j0 + 16] = (ti == tj1) ? v1 : 0.0f;
        }
}

extern "C" void kernel_launch(void* const* d_in, const int* in_sizes, int n_in,
                              void* d_out, int out_size, void* d_ws, size_t ws_size,
                              hipStream_t stream) {
    const float* emb1 = (const float*)d_in[0];
    const float* emb2 = (const float*)d_in[1];
    const int*   t1   = (const int*)d_in[2];
    const int*   t2   = (const int*)d_in[3];
    const float* Wq   = (const float*)d_in[4];
    const float* bq   = (const float*)d_in[5];
    const float* Wk   = (const float*)d_in[6];
    const float* bk   = (const float*)d_in[7];
    float* out = (float*)d_out;

    char* ws = (char*)d_ws;
    const size_t MB = (size_t)1 << 20;
    _Float16* e1h = (_Float16*)(ws);               // 1 MB  (2048x256 f16)
    _Float16* e2h = (_Float16*)(ws + 1 * MB);      // 1 MB
    _Float16* Wqt = (_Float16*)(ws + 2 * MB);      // 1 MB  (2048x256 f16, transposed)
    _Float16* Wkt = (_Float16*)(ws + 3 * MB);      // 1 MB
    _Float16* Q1  = (_Float16*)(ws + 4 * MB);      // 8 MB  (2048x2048 f16)
    _Float16* Q2  = (_Float16*)(ws + 12 * MB);     // 8 MB
    _Float16* K1  = (_Float16*)(ws + 20 * MB);     // 8 MB
    _Float16* K2  = (_Float16*)(ws + 28 * MB);     // 8 MB  -> total 36 MB

    // Stage 0: precision conversion + weight transpose
    cvt_emb_kernel<<<dim3(N_NODES * DHEAD / 256, 1, 2), 256, 0, stream>>>(emb1, emb2, e1h, e2h);
    transpose_w_kernel<<<dim3(PDIM / 32, DHEAD / 32, 2), dim3(32, 32), 0, stream>>>(Wq, Wk, Wqt, Wkt);

    // Stage 1: four projection GEMMs (z selects combo)
    proj_kernel<<<dim3(PDIM / 128, N_NODES / 128, 4), 256, 0, stream>>>(
        e1h, e2h, Wqt, Wkt, bq, bk, Q1, Q2, K1, K2);

    // Stage 2: 16 masked score GEMMs (2 directions x 8 heads)
    attn_kernel<<<dim3(N_NODES / 128, N_NODES / 128, 16), 256, 0, stream>>>(
        Q1, Q2, K1, K2, t1, t2, out);
}
